// GRUAttnDecoderRNN_22600117911849
// MI455X (gfx1250) — matrix operand order repaired and checked by reading, not compiled, not run on hardware
//
#include <hip/hip_runtime.h>
#include <hip/hip_bf16.h>

// ---------------------------------------------------------------------------
// GRU attention decoder step for MI455X (gfx1250, wave32, WMMA).
// Dominant cost: h_new @ Wo.T (16.8 GFLOP, 131MB of fp32 weights streamed
// once -> ~5.6us HBM floor). All GEMMs use v_wmma_f32_16x16x32_bf16 with
// on-the-fly fp32->bf16 conversion and fp32 accumulation.
// ---------------------------------------------------------------------------

#define BB 256
#define LL 50
#define HH 1024
#define VV 32000

typedef __attribute__((ext_vector_type(16))) __bf16 v16bf;
typedef __attribute__((ext_vector_type(8)))  float  v8f;

// ---------------------------------------------------------------------------
// Kernel 1: embedding gather + attention scores + softmax + context vector.
// Produces x_cat = [embedded | attn_applied] (B x 2H) and attn_weights.
// ---------------------------------------------------------------------------
__global__ __launch_bounds__(256) void attn_kernel(
    const int* __restrict__ input, const float* __restrict__ hidden,
    const float* __restrict__ enc, const float* __restrict__ emb,
    const float* __restrict__ Wa, const float* __restrict__ ba,
    float* __restrict__ x_cat, float* __restrict__ attn_out)
{
    const int b = blockIdx.x;
    const int tid = threadIdx.x;
    __shared__ float s_in[2 * HH];   // [embedded | hidden] for the scores
    __shared__ float s_w[64];        // scores -> weights

    const int idx = input[b];
    const float* e  = emb + (size_t)idx * HH;
    const float* hd = hidden + (size_t)b * HH;
    for (int j = tid; j < HH; j += 256) {
        float ev = e[j];
        s_in[j] = ev;
        x_cat[(size_t)b * 2 * HH + j] = ev;  // first half of Wc input
        s_in[HH + j] = hd[j];
    }
    __syncthreads();

    // scores[l] = ba[l] + attn_in . Wa[l]   (only 50 of them, tiny)
    for (int l = tid; l < LL; l += 256) {
        const float* w = Wa + (size_t)l * 2 * HH;
        float acc = ba[l];
        for (int j = 0; j < 2 * HH; ++j) acc += s_in[j] * w[j];
        s_w[l] = acc;
    }
    __syncthreads();

    if (tid == 0) {
        float m = -3.402823e38f;
        for (int l = 0; l < LL; ++l) m = fmaxf(m, s_w[l]);
        float s = 0.f;
        for (int l = 0; l < LL; ++l) { float ev = __expf(s_w[l] - m); s_w[l] = ev; s += ev; }
        float inv = 1.0f / s;
        for (int l = 0; l < LL; ++l) s_w[l] *= inv;
    }
    __syncthreads();

    for (int l = tid; l < LL; l += 256)
        attn_out[(size_t)b * LL + l] = s_w[l];

    // context = weights @ encoder_outputs[b]  -> second half of Wc input
    for (int j = tid; j < HH; j += 256) {
        float acc = 0.f;
        for (int l = 0; l < LL; ++l)
            acc += s_w[l] * enc[((size_t)b * LL + l) * HH + j];
        x_cat[(size_t)b * 2 * HH + HH + j] = acc;
    }
}

// ---------------------------------------------------------------------------
// Generic WMMA GEMM:  C[M x N] = act( A[M x K] * W[N x K]^T + bias[N] )
// fp32 in/out, bf16 WMMA internally, fp32 accumulate.
// One wave handles a 16-row x 64-col strip (4 n-tiles, A-fragment reused 4x).
// M % 16 == 0, N % 64 == 0, K % 32 == 0.  ACT: 0 = none, 1 = relu.
// ---------------------------------------------------------------------------
template<int ACT>
__global__ __launch_bounds__(256) void gemm_bf16_wmma(
    const float* __restrict__ A, const float* __restrict__ W,
    const float* __restrict__ bias, float* __restrict__ C,
    int M, int N, int K)
{
    const int lane = threadIdx.x & 31;
    const int wave = threadIdx.x >> 5;
    const int mt_count = M >> 4;

    const int task    = blockIdx.x * 8 + wave;
    const int m_tile  = task % mt_count;   // m fast-varying: waves share W rows
    const int n_group = task / mt_count;
    const int m0 = m_tile << 4;
    const int n0 = n_group << 6;
    if (n0 >= N) return;                   // wave-uniform: EXEC stays all-1s

    const int hl = lane >> 4;              // half-wave select (0/1)
    const int ll = lane & 15;

    v8f acc0 = {}, acc1 = {}, acc2 = {}, acc3 = {};

    const float* arow = A + (size_t)(m0 + ll) * K;
    for (int kb = 0; kb < K; kb += 32) {
        // A fragment, 16x32 bf16 (contiguous k, matching B):
        // lanes 0-15: elems 0..15 -> K kb+0..15
        // lanes 16-31: elems 0..15 -> K kb+16..31
        v16bf af;
        {
            const float* p0 = arow + kb + hl * 16;
            const float* p1 = arow + kb + hl * 16 + 8;
#pragma unroll
            for (int i = 0; i < 8; ++i) {
                af[i]     = (__bf16)p0[i];
                af[8 + i] = (__bf16)p1[i];
            }
        }
#pragma unroll
        for (int nt = 0; nt < 4; ++nt) {
            // B fragment, 32x16: lane = column n0+nt*16+ll,
            // lanes 0-15 hold K kb+0..15, lanes 16-31 hold K kb+16..31
            const float* wrow = W + (size_t)(n0 + nt * 16 + ll) * K + kb + hl * 16;
            v16bf bf;
#pragma unroll
            for (int i = 0; i < 16; ++i) bf[i] = (__bf16)wrow[i];
            v8f& acc = (nt == 0) ? acc0 : (nt == 1) ? acc1 : (nt == 2) ? acc2 : acc3;
            acc = __builtin_amdgcn_wmma_f32_16x16x32_bf16(
                false, af, false, bf, (short)0, acc, false, false);
        }
    }

    // Epilogue: C/D layout -> lane = column, VGPR r = row r (+8 for hi half)
#pragma unroll
    for (int nt = 0; nt < 4; ++nt) {
        v8f acc = (nt == 0) ? acc0 : (nt == 1) ? acc1 : (nt == 2) ? acc2 : acc3;
        const int col = n0 + nt * 16 + ll;
        const float bv = bias ? bias[col] : 0.0f;
#pragma unroll
        for (int r = 0; r < 8; ++r) {
            const int row = m0 + hl * 8 + r;
            float v = acc[r] + bv;
            if (ACT == 1) v = fmaxf(v, 0.0f);
            C[(size_t)row * N + col] = v;
        }
    }
}

// ---------------------------------------------------------------------------
// Kernel: GRU gate fusion.  h_new = (1-z)*n + z*h
// ---------------------------------------------------------------------------
__global__ __launch_bounds__(256) void gru_gate_kernel(
    const float* __restrict__ xg, const float* __restrict__ hg,
    const float* __restrict__ hidden, float* __restrict__ h_new)
{
    const int i = blockIdx.x * 256 + threadIdx.x;   // over B*H
    const int b = i / HH;
    const int j = i - b * HH;
    const size_t g = (size_t)b * 3 * HH + j;
    const float xr = xg[g], xz = xg[g + HH], xn = xg[g + 2 * HH];
    const float hr = hg[g], hz = hg[g + HH], hn = hg[g + 2 * HH];
    const float r = 1.0f / (1.0f + __expf(-(xr + hr)));
    const float z = 1.0f / (1.0f + __expf(-(xz + hz)));
    const float n = tanhf(xn + r * hn);
    const float h = hidden[(size_t)b * HH + j];
    h_new[(size_t)b * HH + j] = (1.0f - z) * n + z * h;
}

// ---------------------------------------------------------------------------
// Kernel: per-row max + log-sum-exp over V (pass 1 of log_softmax)
// ---------------------------------------------------------------------------
__global__ __launch_bounds__(256) void lse_kernel(
    const float* __restrict__ logits, float* __restrict__ stats)
{
    const int b = blockIdx.x;
    const int tid = threadIdx.x;
    __shared__ float red[256];
    const float* row = logits + (size_t)b * VV;

    float m = -3.402823e38f;
    for (int j = tid; j < VV; j += 256) m = fmaxf(m, row[j]);
    red[tid] = m; __syncthreads();
    for (int s = 128; s > 0; s >>= 1) {
        if (tid < s) red[tid] = fmaxf(red[tid], red[tid + s]);
        __syncthreads();
    }
    m = red[0]; __syncthreads();

    float s = 0.f;
    for (int j = tid; j < VV; j += 256) s += __expf(row[j] - m);
    red[tid] = s; __syncthreads();
    for (int s2 = 128; s2 > 0; s2 >>= 1) {
        if (tid < s2) red[tid] += red[tid + s2];
        __syncthreads();
    }
    if (tid == 0) { stats[b] = m; stats[BB + b] = __logf(red[0]); }
}

// ---------------------------------------------------------------------------
// Kernel: apply log_softmax in place:  out = logit - max - log(sum_exp)
// ---------------------------------------------------------------------------
__global__ __launch_bounds__(256) void lsm_apply_kernel(
    float* __restrict__ logits, const float* __restrict__ stats)
{
    const size_t total = (size_t)BB * VV;
    size_t i = (size_t)blockIdx.x * 256 + threadIdx.x;
    const size_t stride = (size_t)gridDim.x * 256;
    for (; i < total; i += stride) {
        const int b = (int)(i / VV);
        logits[i] = logits[i] - stats[b] - stats[BB + b];
    }
}

// ---------------------------------------------------------------------------
extern "C" void kernel_launch(void* const* d_in, const int* in_sizes, int n_in,
                              void* d_out, int out_size, void* d_ws, size_t ws_size,
                              hipStream_t stream) {
    const int*   input  = (const int*)  d_in[0];
    const float* hidden = (const float*)d_in[1];   // B x H
    const float* enc    = (const float*)d_in[2];   // B x L x H
    const float* emb    = (const float*)d_in[3];   // V x H
    const float* Wa     = (const float*)d_in[4];   // L x 2H
    const float* ba     = (const float*)d_in[5];
    const float* Wc     = (const float*)d_in[6];   // H x 2H
    const float* bc     = (const float*)d_in[7];
    const float* W_ih   = (const float*)d_in[8];   // 3H x H
    const float* b_ih   = (const float*)d_in[9];
    const float* W_hh   = (const float*)d_in[10];  // 3H x H
    const float* b_hh   = (const float*)d_in[11];
    const float* Wo     = (const float*)d_in[12];  // V x H
    const float* bo     = (const float*)d_in[13];

    // d_out = [out: B*V][h_new: B*H][attn_weights: B*L]
    float* logits = (float*)d_out;
    float* h_new  = logits + (size_t)BB * VV;
    float* attn_w = h_new + (size_t)BB * HH;

    // workspace layout (floats)
    float* x_cat = (float*)d_ws;                 // B x 2H
    float* x     = x_cat + (size_t)BB * 2 * HH;  // B x H
    float* xg    = x     + (size_t)BB * HH;      // B x 3H
    float* hg    = xg    + (size_t)BB * 3 * HH;  // B x 3H
    float* stats = hg    + (size_t)BB * 3 * HH;  // 2 * B

    // 1) embedding + attention + context
    attn_kernel<<<BB, 256, 0, stream>>>(input, hidden, enc, emb, Wa, ba, x_cat, attn_w);

    // 2) x = relu(x_cat @ Wc^T + bc)        M=256 N=1024 K=2048
    {
        int blocks = (BB / 16) * (HH / 64) / 8;           // 32
        gemm_bf16_wmma<1><<<blocks, 256, 0, stream>>>(x_cat, Wc, bc, x, BB, HH, 2 * HH);
    }
    // 3) xg = x @ W_ih^T + b_ih             M=256 N=3072 K=1024
    {
        int blocks = (BB / 16) * ((3 * HH) / 64) / 8;     // 96
        gemm_bf16_wmma<0><<<blocks, 256, 0, stream>>>(x, W_ih, b_ih, xg, BB, 3 * HH, HH);
    }
    // 4) hg = h @ W_hh^T + b_hh             M=256 N=3072 K=1024
    {
        int blocks = (BB / 16) * ((3 * HH) / 64) / 8;     // 96
        gemm_bf16_wmma<0><<<blocks, 256, 0, stream>>>(hidden, W_hh, b_hh, hg, BB, 3 * HH, HH);
    }
    // 5) GRU gates -> h_new (directly into its d_out slot)
    gru_gate_kernel<<<(BB * HH) / 256, 256, 0, stream>>>(xg, hg, hidden, h_new);

    // 6) logits = h_new @ Wo^T + bo         M=256 N=32000 K=1024  (dominant GEMM)
    {
        int blocks = (BB / 16) * (VV / 64) / 8;           // 1000
        gemm_bf16_wmma<0><<<blocks, 256, 0, stream>>>(h_new, Wo, bo, logits, BB, VV, HH);
    }
    // 7) + 8) log_softmax over V
    lse_kernel<<<BB, 256, 0, stream>>>(logits, stats);
    lsm_apply_kernel<<<4096, 256, 0, stream>>>(logits, stats);
}
